// EncoderBlock_69011534512843
// MI455X (gfx1250) — matrix operand order repaired and checked by reading, not compiled, not run on hardware
//
#include <hip/hip_runtime.h>
#include <stdint.h>

// ---------------------------------------------------------------------------
// Transformer encoder block for MI455X (gfx1250), bf16 WMMA pipeline.
// B=2 T=2048 D=1024 H=16 HS=64 FF=16384.  ~380 GFLOP, matrix-core bound:
// GEMMs + attention on v_wmma_f32_16x16x32_bf16 (f32 accum); straight
// global->LDS tile copies use global_load_async_to_lds_b128 (ASYNCcnt).
// ---------------------------------------------------------------------------

#define B_  2
#define T_  2048
#define D_  1024
#define H_  16
#define HS_ 64
#define FF_ (D_ * H_)        // 16384
#define D3_ (3 * D_)         // 3072
#define MROWS (B_ * T_)      // 4096

typedef __attribute__((ext_vector_type(16))) __bf16 v16bf;
typedef __attribute__((ext_vector_type(8)))  float  v8f;

union BF16Frag { v16bf v; uint32_t u[8]; };

typedef __attribute__((address_space(3))) unsigned short lds_ushort;

__device__ static inline unsigned short f2bf(float f) {
  union { float f; uint32_t u; } x; x.f = f;
  uint32_t r = x.u + 0x7FFFu + ((x.u >> 16) & 1u);   // round-to-nearest-even
  return (unsigned short)(r >> 16);
}

__device__ static inline v8f wmma_bf16(const v16bf& a, const v16bf& b, const v8f& c) {
  return __builtin_amdgcn_wmma_f32_16x16x32_bf16(false, a, false, b, (short)0, c,
                                                 false, false);
}

// 32-bit LDS byte address of a __shared__ location (addrspace(3) offset).
__device__ static inline uint32_t lds_off(const unsigned short* p) {
  return (uint32_t)(uintptr_t)(lds_ushort*)p;
}

// Async 16-byte global->LDS copy (GVS mode: saddr base + 32-bit voffset).
// Tracked by ASYNCcnt; completes LDS write when asynccnt drains.
__device__ static inline void async_copy_b128(uint32_t lds_addr, const void* base,
                                              uint32_t byte_off) {
  asm volatile("global_load_async_to_lds_b128 %0, %1, %2"
               :
               : "v"(lds_addr), "v"(byte_off), "s"(base)
               : "memory");
}

__device__ static inline void wait_async0() {
#if __has_builtin(__builtin_amdgcn_s_wait_asynccnt)
  __builtin_amdgcn_s_wait_asynccnt(0);
#else
  asm volatile("s_wait_asynccnt 0x0" ::: "memory");
#endif
}

// ---------------------------------------------------------------------------
// fp32 -> bf16 cast (grid-stride)
// ---------------------------------------------------------------------------
__global__ void cast_f32_bf16_kernel(const float* __restrict__ s,
                                     unsigned short* __restrict__ d,
                                     long long n) {
  long long i  = (long long)blockIdx.x * blockDim.x + threadIdx.x;
  long long st = (long long)gridDim.x * blockDim.x;
  for (; i < n; i += st) d[i] = f2bf(s[i]);
}

// ---------------------------------------------------------------------------
// Tiled bf16 GEMM:  C[M,N] = A[M,K] * B[K,N] + bias  (optional ReLU)
// Block = 256 threads (8 waves), tile 64(M) x 128(N), K staged 32 at a time.
// A tile staged with global_load_async_to_lds_b128; B tile pair-packed.
// Each wave computes a 32x32 tile = 2x2 WMMA accumulators.
// ---------------------------------------------------------------------------
template <bool RELU, bool OUT_BF16>
__global__ __launch_bounds__(256)
void gemm_bf16_kernel(const unsigned short* __restrict__ A,
                      const unsigned short* __restrict__ Bm,
                      const float* __restrict__ bias,
                      void* __restrict__ Cout,
                      int M, int N, int K) {
  __shared__ unsigned short As[64][32];       // 4 KB, row-major (K pairs contiguous)
  __shared__ unsigned short Bp[16][128][2];   // 8 KB, (k/2, n, k&1) pair-packed

  const int m0 = blockIdx.y * 64;
  const int n0 = blockIdx.x * 128;

  const int lane = threadIdx.x & 31;
  const int wid  = threadIdx.x >> 5;
  const int wm   = (wid >> 2) * 32;           // 0 or 32
  const int wn   = (wid & 3) * 32;            // 0..96
  const int lh   = lane & 15;
  const int hi   = lane >> 4;                 // lane half: 0/1

  v8f acc[2][2];
#pragma unroll
  for (int mi = 0; mi < 2; ++mi)
#pragma unroll
    for (int nj = 0; nj < 2; ++nj)
#pragma unroll
      for (int e = 0; e < 8; ++e) acc[mi][nj][e] = 0.0f;

  const int arow = threadIdx.x >> 2;
  const int ach  = (threadIdx.x & 3) * 8;
  const uint32_t a_lds = lds_off(&As[arow][ach]);

  for (int k0 = 0; k0 < K; k0 += 32) {
    __syncthreads();   // previous iteration's LDS reads are done

    // ---- A tile 64x32: async DMA straight into LDS (one b128 per thread) ----
    async_copy_b128(a_lds, A,
                    (uint32_t)(((size_t)(m0 + arow) * K + k0 + ach) * 2));

    // ---- B tile 32x128 into pair-packed layout (needs register repack) ----
#pragma unroll
    for (int i = 0; i < 2; ++i) {
      int c   = threadIdx.x * 2 + i;          // 0..511
      int row = c >> 4;                       // k within tile 0..31
      int col = (c & 15) * 8;                 // n within tile
      uint4 d = *reinterpret_cast<const uint4*>(Bm + (size_t)(k0 + row) * N + n0 + col);
      const unsigned short* e = reinterpret_cast<const unsigned short*>(&d);
      int kp = row >> 1, kb = row & 1;
#pragma unroll
      for (int t = 0; t < 8; ++t) Bp[kp][col + t][kb] = e[t];
    }
    wait_async0();      // A tile landed in LDS
    __syncthreads();

    if (k0 + 32 < K)    // gfx1250 global_prefetch of next B slab
      __builtin_prefetch(Bm + (size_t)(k0 + 32) * N + n0, 0, 1);

    // ---- fragments ----
    BF16Frag aF[2], bF[2];
#pragma unroll
    for (int mi = 0; mi < 2; ++mi) {
      int m = wm + mi * 16 + lh;
#pragma unroll
      for (int i = 0; i < 8; ++i) {
        int kk = hi * 16 + 2 * i;
        aF[mi].u[i] = *reinterpret_cast<const uint32_t*>(&As[m][kk]);
      }
    }
#pragma unroll
    for (int nj = 0; nj < 2; ++nj) {
      int n = wn + nj * 16 + lh;
#pragma unroll
      for (int i = 0; i < 8; ++i) {
        int kp = hi * 8 + i;
        bF[nj].u[i] = *reinterpret_cast<const uint32_t*>(&Bp[kp][n][0]);
      }
    }
#pragma unroll
    for (int mi = 0; mi < 2; ++mi)
#pragma unroll
      for (int nj = 0; nj < 2; ++nj)
        acc[mi][nj] = wmma_bf16(aF[mi].v, bF[nj].v, acc[mi][nj]);
  }

  // ---- epilogue: bias (+ReLU), store f32 or bf16 ----
#pragma unroll
  for (int mi = 0; mi < 2; ++mi)
#pragma unroll
    for (int nj = 0; nj < 2; ++nj) {
      int n   = n0 + wn + nj * 16 + lh;
      float bv = bias[n];
#pragma unroll
      for (int j = 0; j < 8; ++j) {
        int mrow = m0 + wm + mi * 16 + j + hi * 8;
        float v = acc[mi][nj][j] + bv;
        if (RELU) v = fmaxf(v, 0.0f);
        if (OUT_BF16)
          ((unsigned short*)Cout)[(size_t)mrow * N + n] = f2bf(v);
        else
          ((float*)Cout)[(size_t)mrow * N + n] = v;
      }
    }
}

// ---------------------------------------------------------------------------
// Flash attention (causal).  One block (4 waves) per (b, h, 64-row q tile).
// Wave w owns 16 q rows; K tile async-DMA'd to LDS, V pair-packed; online
// softmax; both QK^T and P*V on v_wmma_f32_16x16x32_bf16.
// ---------------------------------------------------------------------------
__global__ __launch_bounds__(128)
void attn_kernel(const unsigned short* __restrict__ qkv,
                 unsigned short* __restrict__ yout) {
  __shared__ unsigned short Kt[64 * 64];        // 8 KB row-major (kcol, d)
  __shared__ unsigned short Vp[32 * 64 * 2];    // 8 KB pair-packed (k/2, d, k&1)
  __shared__ unsigned short Pl[4][16 * 64];     // 8 KB per-wave P tiles

  const int bid = blockIdx.x;
  const int qt  = bid & (T_ / 64 - 1);          // 0..31
  const int h   = (bid >> 5) & (H_ - 1);
  const int b   = bid >> 9;

  const int lane = threadIdx.x & 31;
  const int wid  = threadIdx.x >> 5;            // 0..3
  const int lh   = lane & 15;
  const int hi   = lane >> 4;
  const int qbase = qt * 64 + wid * 16;

  // ---- Q fragments: 16 rows x 64 d, two K=32 chunks, loaded from global ----
  BF16Frag qF[2];
#pragma unroll
  for (int ch = 0; ch < 2; ++ch) {
    int q = qbase + lh;
#pragma unroll
    for (int i = 0; i < 8; ++i) {
      int kk = hi * 16 + 2 * i + ch * 32;
      size_t g = (size_t)(b * T_ + q) * D3_ + h * HS_ + kk;
      qF[ch].u[i] = *reinterpret_cast<const uint32_t*>(qkv + g);
    }
  }

  v8f o[4];
  float mi[8], li[8];
#pragma unroll
  for (int t = 0; t < 4; ++t)
#pragma unroll
    for (int e = 0; e < 8; ++e) o[t][e] = 0.0f;
#pragma unroll
  for (int j = 0; j < 8; ++j) { mi[j] = -__builtin_inff(); li[j] = 0.0f; }

  for (int kt = 0; kt <= qt; ++kt) {
    __syncthreads();
    // ---- stage K (async DMA, row-major) and V (pair-packed) tiles 64x64 ----
#pragma unroll
    for (int i = 0; i < 4; ++i) {
      int c   = threadIdx.x * 4 + i;            // 0..511
      int row = c >> 3;                          // k position 0..63
      int col = (c & 7) * 8;                     // d offset
      size_t gk = (size_t)(b * T_ + kt * 64 + row) * D3_ + D_ + h * HS_ + col;
      async_copy_b128(lds_off(&Kt[row * 64 + col]), qkv, (uint32_t)(gk * 2));
      uint4 dv = *reinterpret_cast<const uint4*>(qkv + gk + D_);  // V section
      const unsigned short* e = reinterpret_cast<const unsigned short*>(&dv);
      int kp = row >> 1, kb = row & 1;
#pragma unroll
      for (int t = 0; t < 8; ++t) Vp[(kp * 64 + col + t) * 2 + kb] = e[t];
    }
    wait_async0();
    __syncthreads();

    // ---- S = Q K^T : 4 C tiles (16q x 16k), 2 WMMAs each ----
    v8f s[4];
#pragma unroll
    for (int nj = 0; nj < 4; ++nj) {
#pragma unroll
      for (int e = 0; e < 8; ++e) s[nj][e] = 0.0f;
      int n = nj * 16 + lh;                      // k col within tile
#pragma unroll
      for (int ch = 0; ch < 2; ++ch) {
        BF16Frag kF;
#pragma unroll
        for (int i = 0; i < 8; ++i) {
          int d = ch * 32 + hi * 16 + 2 * i;
          kF.u[i] = *reinterpret_cast<const uint32_t*>(&Kt[n * 64 + d]);
        }
        s[nj] = wmma_bf16(qF[ch].v, kF.v, s[nj]);
      }
    }

    // ---- online softmax (rows j + hi*8; 16-lane reductions stay in half) ----
    const bool diag = (kt == qt);
#pragma unroll
    for (int j = 0; j < 8; ++j) {
      int q = qbase + j + hi * 8;
      float sv[4];
      float rm = -__builtin_inff();
#pragma unroll
      for (int nj = 0; nj < 4; ++nj) {
        int col = kt * 64 + nj * 16 + lh;
        float xv = s[nj][j] * 0.125f;            // 1/sqrt(64)
        if (diag && col > q) xv = -__builtin_inff();
        sv[nj] = xv;
        rm = fmaxf(rm, xv);
      }
#pragma unroll
      for (int msk = 1; msk < 16; msk <<= 1) rm = fmaxf(rm, __shfl_xor(rm, msk, 32));
      float mnew  = fmaxf(mi[j], rm);
      float alpha = __expf(mi[j] - mnew);
      float rs = 0.0f;
#pragma unroll
      for (int nj = 0; nj < 4; ++nj) {
        float p = __expf(sv[nj] - mnew);
        rs += p;
        Pl[wid][(j + hi * 8) * 64 + nj * 16 + lh] = f2bf(p);
      }
#pragma unroll
      for (int msk = 1; msk < 16; msk <<= 1) rs += __shfl_xor(rs, msk, 32);
      li[j] = alpha * li[j] + rs;
      mi[j] = mnew;
#pragma unroll
      for (int t = 0; t < 4; ++t) o[t][j] *= alpha;
    }

    // ---- O += P V : P from per-wave LDS (A layout), V pair-packed (B layout)
    BF16Frag pF[2];
#pragma unroll
    for (int ch = 0; ch < 2; ++ch)
#pragma unroll
      for (int i = 0; i < 8; ++i) {
        int kk = hi * 16 + 2 * i + ch * 32;
        pF[ch].u[i] = *reinterpret_cast<const uint32_t*>(&Pl[wid][lh * 64 + kk]);
      }
#pragma unroll
    for (int t = 0; t < 4; ++t) {
      int d = t * 16 + lh;
#pragma unroll
      for (int ch = 0; ch < 2; ++ch) {
        BF16Frag vF;
#pragma unroll
        for (int i = 0; i < 8; ++i) {
          int kp = ch * 16 + hi * 8 + i;
          vF.u[i] = *reinterpret_cast<const uint32_t*>(&Vp[(kp * 64 + d) * 2]);
        }
        o[t] = wmma_bf16(pF[ch].v, vF.v, o[t]);
      }
    }
  }

  // ---- normalize and store bf16 into y[b, t, h*64 + d] ----
#pragma unroll
  for (int t = 0; t < 4; ++t) {
    int d = h * HS_ + t * 16 + lh;
#pragma unroll
    for (int j = 0; j < 8; ++j) {
      int q = qbase + j + hi * 8;
      float v = o[t][j] / li[j];
      yout[(size_t)(b * T_ + q) * D_ + d] = f2bf(v);
    }
  }
}

// ---------------------------------------------------------------------------
// out = LayerNorm(a + r) * g + beta ;  f32 out, optional bf16 mirror.
// One block per row (D=1024), 256 threads x 4 elements.
// ---------------------------------------------------------------------------
__global__ __launch_bounds__(256)
void resid_ln_kernel(const float* __restrict__ a, const float* __restrict__ r,
                     const float* __restrict__ g, const float* __restrict__ beta,
                     float* __restrict__ of, unsigned short* __restrict__ ob) {
  __shared__ float red[2][8];
  const size_t row = blockIdx.x;
  const float* pa = a + row * D_;
  const float* pr = r + row * D_;
  float v[4], s = 0.0f, s2 = 0.0f;
#pragma unroll
  for (int i = 0; i < 4; ++i) {
    int c = threadIdx.x + i * 256;
    v[i] = pa[c] + pr[c];
    s += v[i];
    s2 += v[i] * v[i];
  }
#pragma unroll
  for (int msk = 1; msk < 32; msk <<= 1) {
    s  += __shfl_xor(s, msk, 32);
    s2 += __shfl_xor(s2, msk, 32);
  }
  int wid = threadIdx.x >> 5, lane = threadIdx.x & 31;
  if (lane == 0) { red[0][wid] = s; red[1][wid] = s2; }
  __syncthreads();
  float ts = 0.0f, ts2 = 0.0f;
#pragma unroll
  for (int w = 0; w < 8; ++w) { ts += red[0][w]; ts2 += red[1][w]; }
  float mean = ts * (1.0f / D_);
  float var  = ts2 * (1.0f / D_) - mean * mean;
  float inv  = rsqrtf(var + 1e-5f);
#pragma unroll
  for (int i = 0; i < 4; ++i) {
    int c = threadIdx.x + i * 256;
    float y = (v[i] - mean) * inv * g[c] + beta[c];
    of[row * D_ + c] = y;
    if (ob) ob[row * D_ + c] = f2bf(y);
  }
}

// ---------------------------------------------------------------------------
extern "C" void kernel_launch(void* const* d_in, const int* in_sizes, int n_in,
                              void* d_out, int out_size, void* d_ws, size_t ws_size,
                              hipStream_t stream) {
  (void)in_sizes; (void)n_in; (void)out_size; (void)ws_size;
  const float* x      = (const float*)d_in[0];
  const float* w_qkv  = (const float*)d_in[1];
  const float* b_qkv  = (const float*)d_in[2];
  const float* w_fc   = (const float*)d_in[3];
  const float* b_fc   = (const float*)d_in[4];
  const float* g1     = (const float*)d_in[5];
  const float* beta1  = (const float*)d_in[6];
  const float* w_ff1  = (const float*)d_in[7];
  const float* b_ff1  = (const float*)d_in[8];
  const float* w_ff2  = (const float*)d_in[9];
  const float* b_ff2  = (const float*)d_in[10];
  const float* g2     = (const float*)d_in[11];
  const float* beta2  = (const float*)d_in[12];

  size_t off = 0;
  auto carve = [&](size_t elems, size_t esz) -> void* {
    void* p = (char*)d_ws + off;
    off += (elems * esz + 255) & ~(size_t)255;
    return p;
  };
  unsigned short* xb    = (unsigned short*)carve((size_t)MROWS * D_, 2);
  unsigned short* wqb   = (unsigned short*)carve((size_t)D_ * D3_, 2);
  unsigned short* wfcb  = (unsigned short*)carve((size_t)D_ * D_, 2);
  unsigned short* wf1b  = (unsigned short*)carve((size_t)D_ * FF_, 2);
  unsigned short* wf2b  = (unsigned short*)carve((size_t)FF_ * D_, 2);
  unsigned short* qkvb  = (unsigned short*)carve((size_t)MROWS * D3_, 2);
  unsigned short* yb    = (unsigned short*)carve((size_t)MROWS * D_, 2);
  float*          projf = (float*)carve((size_t)MROWS * D_, 4);
  float*          h1f   = (float*)carve((size_t)MROWS * D_, 4);
  unsigned short* h1b   = (unsigned short*)carve((size_t)MROWS * D_, 2);
  unsigned short* f1b   = (unsigned short*)carve((size_t)MROWS * FF_, 2);
  float*          f2f   = (float*)carve((size_t)MROWS * D_, 4);

  // 1) one-time bf16 casts of activations + weights
  cast_f32_bf16_kernel<<<2048, 256, 0, stream>>>(x, xb, (long long)MROWS * D_);
  cast_f32_bf16_kernel<<<2048, 256, 0, stream>>>(w_qkv, wqb, (long long)D_ * D3_);
  cast_f32_bf16_kernel<<<2048, 256, 0, stream>>>(w_fc, wfcb, (long long)D_ * D_);
  cast_f32_bf16_kernel<<<2048, 256, 0, stream>>>(w_ff1, wf1b, (long long)D_ * FF_);
  cast_f32_bf16_kernel<<<2048, 256, 0, stream>>>(w_ff2, wf2b, (long long)FF_ * D_);

  // 2) qkv = x @ w_qkv + b_qkv        [4096 x 3072]
  gemm_bf16_kernel<false, true><<<dim3(D3_ / 128, MROWS / 64), 256, 0, stream>>>(
      xb, wqb, b_qkv, qkvb, MROWS, D3_, D_);

  // 3) flash attention -> y (bf16)
  attn_kernel<<<dim3(B_ * H_ * (T_ / 64)), 128, 0, stream>>>(qkvb, yb);

  // 4) proj = y @ w_fc + b_fc (f32)
  gemm_bf16_kernel<false, false><<<dim3(D_ / 128, MROWS / 64), 256, 0, stream>>>(
      yb, wfcb, b_fc, projf, MROWS, D_, D_);

  // 5) h1 = LN(x + proj)
  resid_ln_kernel<<<dim3(MROWS), 256, 0, stream>>>(x, projf, g1, beta1, h1f, h1b);

  // 6) ff1 = relu(h1 @ w_ff1 + b_ff1)  [4096 x 16384] bf16
  gemm_bf16_kernel<true, true><<<dim3(FF_ / 128, MROWS / 64), 256, 0, stream>>>(
      h1b, wf1b, b_ff1, f1b, MROWS, FF_, D_);

  // 7) ff2 = ff1 @ w_ff2 + b_ff2 (f32), K = 16384
  gemm_bf16_kernel<false, false><<<dim3(D_ / 128, MROWS / 64), 256, 0, stream>>>(
      f1b, wf2b, b_ff2, f2f, MROWS, D_, FF_);

  // 8) out = LN(h1 + ff2)
  resid_ln_kernel<<<dim3(MROWS), 256, 0, stream>>>(h1f, f2f, g2, beta2,
                                                   (float*)d_out, nullptr);
}
